// Attention_76081050682051
// MI455X (gfx1250) — matrix operand order repaired and checked by reading, not compiled, not run on hardware
//
#include <hip/hip_runtime.h>
#include <hip/hip_bf16.h>

// ---- problem constants (from reference setup_inputs) ----
constexpr int B_  = 4;
constexpr int S_  = 2048;
constexpr int E_  = 256;    // embed == head dim F
constexpr int H_  = 8;
constexpr int HE_ = H_ * E_;      // 2048
constexpr int BS_ = B_ * S_;      // 8192
constexpr int BH_ = B_ * H_;      // 32

typedef __attribute__((ext_vector_type(16))) _Float16 v16h;
typedef __attribute__((ext_vector_type(8)))  _Float16 v8h;
typedef __attribute__((ext_vector_type(8)))  float    v8f;

// ---------------- WMMA fragment helpers (wave32, 16x16x32 f16) -------------
// A (16x32 f16): lane L -> row m=L&15; halves i = K (L>>4)*16 + i (same order as B).
// B (32x16 f16): lane L -> col n=L&15; halves i = K (L>>4)*16 + i.
// C/D (16x16 f32): lane L -> col n=L&15; VGPR i -> row m = i + 8*(L>>4).
__device__ inline v16h pack16(v8h lo, v8h hi) {
  v16h r;
#pragma unroll
  for (int j = 0; j < 8; ++j) { r[j] = lo[j]; r[8 + j] = hi[j]; }
  return r;
}
// p = rowbase + kk*32 + (lane>>4)*16 ; 16 contiguous halves
__device__ inline v16h ldfragA(const _Float16* p) {
  return pack16(*(const v8h*)p, *(const v8h*)(p + 8));
}
// p = 16 contiguous halves along contraction dim
__device__ inline v16h ldfragB(const _Float16* p) {
  return pack16(*(const v8h*)p, *(const v8h*)(p + 8));
}
__device__ inline v8f wmma16(v16h a, v16h b, v8f c) {
  return __builtin_amdgcn_wmma_f32_16x16x32_f16(false, a, false, b,
                                                (short)0, c, false, false);
}

// ---- CDNA5 async copy: global -> LDS, 16B per lane, tracked by ASYNCcnt ---
__device__ inline void async_cp16(_Float16* lds, const _Float16* g) {
  const unsigned lofs = (unsigned)(unsigned long long)lds;  // LDS offset in low 32b
  asm volatile("global_load_async_to_lds_b128 %0, %1, off"
               :: "v"(lofs), "v"(g) : "memory");
}
__device__ inline void wait_async16() {   // <=16 outstanding (next tile in flight)
  asm volatile("s_wait_asynccnt 0x10" ::: "memory");
}
__device__ inline void wait_async0() {
  asm volatile("s_wait_asynccnt 0x0" ::: "memory");
}

// ---------------- 1) LayerNorm + f16 cast:  xn[b,s,e] ----------------------
__global__ __launch_bounds__(256) void ln_cast_kernel(const float* __restrict__ x,
                                                      const float* __restrict__ scale,
                                                      _Float16* __restrict__ xn) {
  const int row = blockIdx.x;            // 0..BS_-1
  const int t   = threadIdx.x;           // 0..255 (== e)
  const float v = x[row * E_ + t];
  float sum = v, sq = v * v;
#pragma unroll
  for (int o = 16; o > 0; o >>= 1) {
    sum += __shfl_xor(sum, o, 32);
    sq  += __shfl_xor(sq,  o, 32);
  }
  __shared__ float s1[8], s2[8];
  if ((t & 31) == 0) { s1[t >> 5] = sum; s2[t >> 5] = sq; }
  __syncthreads();
  float tot = 0.f, tot2 = 0.f;
#pragma unroll
  for (int i = 0; i < 8; ++i) { tot += s1[i]; tot2 += s2[i]; }
  const float mu  = tot * (1.0f / E_);
  const float var = tot2 * (1.0f / E_) - mu * mu;
  const float r   = rsqrtf(var + 1e-5f);
  xn[row * E_ + t] = (_Float16)((v - mu) * r * scale[t]);
}

// ---------------- 2) weight casts ------------------------------------------
// Wq/Wk/Wv: [H,E,E] (e,f) -> f16 transposed per head: Wt[h][f][e]
__global__ __launch_bounds__(256) void cast_qkv_kernel(const float* __restrict__ Wq,
                                                       const float* __restrict__ Wk,
                                                       const float* __restrict__ Wv,
                                                       _Float16* __restrict__ Wqt,
                                                       _Float16* __restrict__ Wkt,
                                                       _Float16* __restrict__ Wvt) {
  const int idx = blockIdx.x * 256 + threadIdx.x;      // over 3*H*E*E
  const int mat = idx / (H_ * E_ * E_);
  const int rem = idx % (H_ * E_ * E_);
  const int h = rem / (E_ * E_);
  const int e = (rem / E_) % E_;
  const int f = rem % E_;
  const float* src = (mat == 0) ? Wq : (mat == 1) ? Wk : Wv;
  _Float16*   dst = (mat == 0) ? Wqt : (mat == 1) ? Wkt : Wvt;
  dst[(h * E_ + f) * E_ + e] = (_Float16)src[rem];
}

// W_o: [E, H*E] -> f16, same layout (row-major is already B-friendly)
__global__ __launch_bounds__(256) void cast_wo_kernel(const float* __restrict__ Wo,
                                                      _Float16* __restrict__ Wo16) {
  const int idx = blockIdx.x * 256 + threadIdx.x;      // over E_*HE_
  Wo16[idx] = (_Float16)Wo[idx];
}

// ---------------- 3) fused Q/K/V projection --------------------------------
// grid (S/16, H, B), 128 threads = 4 waves; wave w covers f in [w*64, w*64+64)
// Q,K row-major f16 [BH, S, E]; V stored transposed: Vt [BH, E, S]
__global__ __launch_bounds__(128) void qkv_proj_kernel(const _Float16* __restrict__ xn,
                                                       const _Float16* __restrict__ Wqt,
                                                       const _Float16* __restrict__ Wkt,
                                                       const _Float16* __restrict__ Wvt,
                                                       _Float16* __restrict__ Qo,
                                                       _Float16* __restrict__ Ko,
                                                       _Float16* __restrict__ Vt) {
  const int stile = blockIdx.x, h = blockIdx.y, b = blockIdx.z;
  const int wave = threadIdx.x >> 5, lane = threadIdx.x & 31;
  const int m16 = lane & 15, hi = lane >> 4;
  const int sbase = stile * 16;
  const int bh = b * H_ + h;
  const _Float16* A  = xn + (size_t)(b * S_ + sbase) * E_;
  const _Float16* Bq = Wqt + (size_t)h * E_ * E_;   // [f][e]
  const _Float16* Bk = Wkt + (size_t)h * E_ * E_;
  const _Float16* Bv = Wvt + (size_t)h * E_ * E_;
  const int fbase = wave * 64;

  v8f cq[4] = {}, ck[4] = {}, cv[4] = {};
#pragma unroll
  for (int kk = 0; kk < 8; ++kk) {
    const v16h a = ldfragA(A + m16 * E_ + kk * 32 + hi * 16);
#pragma unroll
    for (int nt = 0; nt < 4; ++nt) {
      const int f0 = fbase + nt * 16;
      const size_t boff = (size_t)(f0 + m16) * E_ + kk * 32 + hi * 16;
      cq[nt] = wmma16(a, ldfragB(Bq + boff), cq[nt]);
      ck[nt] = wmma16(a, ldfragB(Bk + boff), ck[nt]);
      cv[nt] = wmma16(a, ldfragB(Bv + boff), cv[nt]);
    }
  }
  _Float16* qdst = Qo + (size_t)(bh * S_ + sbase) * E_;
  _Float16* kdst = Ko + (size_t)(bh * S_ + sbase) * E_;
  _Float16* vdst = Vt + (size_t)bh * E_ * S_;
#pragma unroll
  for (int nt = 0; nt < 4; ++nt) {
    const int f0 = fbase + nt * 16 + m16;
#pragma unroll
    for (int i = 0; i < 8; ++i) {
      const int r = i + 8 * hi;
      qdst[(size_t)r * E_ + f0] = (_Float16)cq[nt][i];
      kdst[(size_t)r * E_ + f0] = (_Float16)ck[nt][i];
      vdst[(size_t)f0 * S_ + sbase + r] = (_Float16)cv[nt][i];  // transposed
    }
  }
}

// ---------------- 4) q2 / k2 row sums of squares ---------------------------
__global__ __launch_bounds__(256) void sumsq_kernel(const _Float16* __restrict__ Q,
                                                    const _Float16* __restrict__ K,
                                                    float* __restrict__ q2,
                                                    float* __restrict__ k2) {
  const int gw   = (blockIdx.x * 256 + threadIdx.x) >> 5;  // 0 .. 2*BH*S-1
  const int lane = threadIdx.x & 31;
  int row = gw;
  const _Float16* src = Q;
  float* dst = q2;
  if (row >= BH_ * S_) { row -= BH_ * S_; src = K; dst = k2; }
  const v8h v = *(const v8h*)(src + (size_t)row * E_ + lane * 8);
  float s = 0.f;
#pragma unroll
  for (int j = 0; j < 8; ++j) { const float f = (float)v[j]; s += f * f; }
#pragma unroll
  for (int o = 16; o > 0; o >>= 1) s += __shfl_xor(s, o, 32);
  if (lane == 0) dst[row] = s;
}

// ---------------- 5) fused RBF attention, async double-buffered ------------
// grid (S/32, H, B), 128 threads = 4 waves.
// wave = (qt<<1)|wf : qt in {0,1} -> query sub-tile, wf in {0,1} -> f-half.
// K-tile (32x256 f16, 16KB) and V-tile (256x32 f16, 16KB) are staged to LDS
// with GLOBAL_LOAD_ASYNC_TO_LDS_B128, double-buffered, shared by all 4 waves.
__global__ __launch_bounds__(128) void attention_kernel(const _Float16* __restrict__ Q,
                                                        const _Float16* __restrict__ K,
                                                        const _Float16* __restrict__ Vt,
                                                        const float* __restrict__ q2,
                                                        const float* __restrict__ k2,
                                                        const float* __restrict__ gamma,
                                                        _Float16* __restrict__ O) {
  const int qgrp = blockIdx.x, h = blockIdx.y, b = blockIdx.z;
  const int tid = threadIdx.x;
  const int wave = tid >> 5, lane = tid & 31;
  const int qt = wave >> 1, wf = wave & 1;
  const int m16 = lane & 15, hi = lane >> 4;
  const int bh = b * H_ + h;
  const int qbase = qgrp * 32 + qt * 16;
  const int fhalf = wf * 128;
  const float g = gamma[h];

  __shared__ _Float16 ldsK[2][32 * 256];   // [key][f]   2 x 16 KB
  __shared__ _Float16 ldsV[2][256 * 32];   // [f][key]   2 x 16 KB
  __shared__ _Float16 ldsP[4][16 * 32];    // per-wave P transpose bounce
  _Float16* myp = ldsP[wave];

  const _Float16* Kbase  = K  + (size_t)bh * S_ * E_;
  const _Float16* Vtbase = Vt + (size_t)bh * E_ * S_;

  // resident Q fragments: full F=256 contraction (8 frags, 64 VGPRs)
  const _Float16* Qrow = Q + (size_t)(bh * S_ + qbase) * E_;
  v16h qf[8];
#pragma unroll
  for (int kk = 0; kk < 8; ++kk)
    qf[kk] = ldfragA(Qrow + m16 * E_ + kk * 32 + hi * 16);

  const v8f q2v = *(const v8f*)(q2 + (size_t)bh * S_ + qbase + hi * 8);

  // async stage one 32-key tile: 8 iters x 128 thr x 16B for K, same for V
  auto issue_tile = [&](int t, int buf) {
    const int key0 = t * 32;
    const _Float16* gK = Kbase + (size_t)key0 * E_;   // 8192 contiguous halves
#pragma unroll
    for (int it = 0; it < 8; ++it) {
      const int ch = it * 128 + tid;                  // 0..1023, 8 halves each
      async_cp16(&ldsK[buf][ch * 8], gK + ch * 8);
      const int f = ch >> 2, so = (ch & 3) * 8;
      async_cp16(&ldsV[buf][f * 32 + so], Vtbase + (size_t)f * S_ + key0 + so);
    }
  };

  constexpr int NT = S_ / 32;   // 64 key tiles
  v8f o[8] = {};
  issue_tile(0, 0);
  for (int t = 0; t < NT; ++t) {
    const int cur = t & 1;
    if (t + 1 < NT) { issue_tile(t + 1, cur ^ 1); wait_async16(); }
    else            { wait_async0(); }
    __syncthreads();                       // staged tile visible to all waves

    const _Float16* Kt = ldsK[cur];
    const _Float16* Vtile = ldsV[cur];
    const int key0 = t * 32;

    // ---- scores: two 16x16 tiles over keys [key0, key0+32), from LDS ----
    v8f c0 = {}, c1 = {};
#pragma unroll
    for (int kk = 0; kk < 8; ++kk) {
      c0 = wmma16(qf[kk], ldfragB(Kt + m16 * 256 + kk * 32 + hi * 16), c0);
      c1 = wmma16(qf[kk], ldfragB(Kt + (16 + m16) * 256 + kk * 32 + hi * 16), c1);
    }
    // ---- elementwise RBF, write P tile (16x32 f16) to LDS ----
    const float k2a = k2[(size_t)bh * S_ + key0 + m16];
    const float k2b = k2[(size_t)bh * S_ + key0 + 16 + m16];
#pragma unroll
    for (int i = 0; i < 8; ++i) {
      const int r = i + 8 * hi;
      float d0 = fmaxf(q2v[i] + k2a - 2.0f * c0[i], 0.0f);
      float d1 = fmaxf(q2v[i] + k2b - 2.0f * c1[i], 0.0f);
      myp[r * 32 + m16]      = (_Float16)__expf(-g * d0);
      myp[r * 32 + 16 + m16] = (_Float16)__expf(-g * d1);
    }
    // intra-wave LDS RAW -> compiler inserts s_wait_dscnt; no barrier needed
    const v16h pa = pack16(*(const v8h*)(myp + m16 * 32 + hi * 16),
                           *(const v8h*)(myp + m16 * 32 + hi * 16 + 8));
    // ---- O += P @ V over this wave's 128 f-columns, V from LDS ----
#pragma unroll
    for (int nt = 0; nt < 8; ++nt) {
      const int f0 = fhalf + nt * 16;
      o[nt] = wmma16(pa, ldfragB(Vtile + (f0 + m16) * 32 + hi * 16), o[nt]);
    }
    __syncthreads();                       // done reading before next overwrite
  }
  // store as f16 into [B,S,H*E] for the output projection
  _Float16* dst = O + ((size_t)b * S_ + qbase) * HE_ + h * E_;
#pragma unroll
  for (int nt = 0; nt < 8; ++nt) {
    const int f0 = fhalf + nt * 16 + m16;
#pragma unroll
    for (int i = 0; i < 8; ++i) {
      const int r = i + 8 * hi;
      dst[(size_t)r * HE_ + f0] = (_Float16)o[nt][i];
    }
  }
}

// ---------------- 6) output projection: [BS,HE] @ Wo^T -> [BS,E] fp32 ------
__global__ __launch_bounds__(32) void out_proj_kernel(const _Float16* __restrict__ A,
                                                      const _Float16* __restrict__ Wo16,
                                                      float* __restrict__ out) {
  const int rbase = blockIdx.x * 16;   // 512 row tiles
  const int nbase = blockIdx.y * 16;   // 16 col tiles
  const int lane = threadIdx.x & 31;
  const int m16 = lane & 15, hi = lane >> 4;
  v8f c = {};
#pragma unroll 4
  for (int kk = 0; kk < HE_ / 32; ++kk) {
    if (kk + 4 < HE_ / 32)
      __builtin_prefetch(A + (size_t)(rbase + m16) * HE_ + (kk + 4) * 32, 0, 1);
    const v16h a = ldfragA(A + (size_t)(rbase + m16) * HE_ + kk * 32 + hi * 16);
    // B[k][n] = Wo[n][k] : row-major [E, HE] -> contiguous along k
    const v16h bb = ldfragB(Wo16 + (size_t)(nbase + m16) * HE_ + kk * 32 + hi * 16);
    c = wmma16(a, bb, c);
  }
#pragma unroll
  for (int i = 0; i < 8; ++i)
    out[(size_t)(rbase + i + 8 * hi) * E_ + nbase + m16] = c[i];
}

// ---------------------------------------------------------------------------
extern "C" void kernel_launch(void* const* d_in, const int* in_sizes, int n_in,
                              void* d_out, int out_size, void* d_ws, size_t ws_size,
                              hipStream_t stream) {
  // inputs: 0:x 1:e(unused) 2:p(unused) 3:ln_scale 4:W_q 5:W_k 6:W_v 7:W_o 8:gamma
  const float* x        = (const float*)d_in[0];
  const float* ln_scale = (const float*)d_in[3];
  const float* W_q      = (const float*)d_in[4];
  const float* W_k      = (const float*)d_in[5];
  const float* W_v      = (const float*)d_in[6];
  const float* W_o      = (const float*)d_in[7];
  const float* gamma    = (const float*)d_in[8];
  float* out = (float*)d_out;

  // ---- workspace carve (~137 MB) ----
  char* w = (char*)d_ws;
  auto take = [&](size_t bytes) { char* p = w; w += (bytes + 255) & ~(size_t)255; return p; };
  _Float16* xn   = (_Float16*)take((size_t)BS_ * E_ * 2);         //   4 MB
  _Float16* Wqt  = (_Float16*)take((size_t)H_ * E_ * E_ * 2);     //   1 MB
  _Float16* Wkt  = (_Float16*)take((size_t)H_ * E_ * E_ * 2);     //   1 MB
  _Float16* Wvt  = (_Float16*)take((size_t)H_ * E_ * E_ * 2);     //   1 MB
  _Float16* Wo16 = (_Float16*)take((size_t)E_ * HE_ * 2);         //   1 MB
  _Float16* Q    = (_Float16*)take((size_t)BH_ * S_ * E_ * 2);    //  32 MB
  _Float16* K    = (_Float16*)take((size_t)BH_ * S_ * E_ * 2);    //  32 MB
  _Float16* Vt   = (_Float16*)take((size_t)BH_ * E_ * S_ * 2);    //  32 MB
  float*    q2   = (float*)   take((size_t)BH_ * S_ * 4);         // 256 KB
  float*    k2   = (float*)   take((size_t)BH_ * S_ * 4);         // 256 KB
  _Float16* AO   = (_Float16*)take((size_t)BS_ * HE_ * 2);        //  32 MB

  ln_cast_kernel<<<BS_, 256, 0, stream>>>(x, ln_scale, xn);
  cast_qkv_kernel<<<(3 * H_ * E_ * E_) / 256, 256, 0, stream>>>(W_q, W_k, W_v, Wqt, Wkt, Wvt);
  cast_wo_kernel<<<(E_ * HE_) / 256, 256, 0, stream>>>(W_o, Wo16);
  qkv_proj_kernel<<<dim3(S_ / 16, H_, B_), 128, 0, stream>>>(xn, Wqt, Wkt, Wvt, Q, K, Vt);
  sumsq_kernel<<<(2 * BH_ * S_) / 8, 256, 0, stream>>>(Q, K, q2, k2);
  attention_kernel<<<dim3(S_ / 32, H_, B_), 128, 0, stream>>>(Q, K, Vt, q2, k2, gamma, AO);
  out_proj_kernel<<<dim3(BS_ / 16, E_ / 16), 32, 0, stream>>>(AO, Wo16, out);
}
